// SA_Layer_1606317769130
// MI455X (gfx1250) — matrix operand order repaired and checked by reading, not compiled, not run on hardware
//
#include <hip/hip_runtime.h>
#include <stdint.h>

// Problem constants (match the reference)
#define B_ 8
#define S_ 4096
#define E_ 128
#define H_ 64
#define O_ 64

#define QTILE 128   // queries per block (8 waves x 16 rows)
#define KTILE 64    // keys per LDS tile
#define KP    35    // K/V LDS row stride in dwords (odd -> conflict-free frag reads)
#define PSTR  70    // P scratch row stride in bf16 (140B, 4B-aligned pair reads)
#define NIT   (S_ / KTILE)

#if defined(__has_builtin)
#  if __has_builtin(__builtin_amdgcn_tensor_load_to_lds)
#    define HAVE_TDM 1
#  endif
#endif
#ifndef HAVE_TDM
#  define HAVE_TDM 0
#endif

typedef __attribute__((ext_vector_type(16))) __bf16    v16bf;
typedef __attribute__((ext_vector_type(8)))  float     v8f;
typedef __attribute__((ext_vector_type(4)))  uint32_t  v4u;
typedef __attribute__((ext_vector_type(8)))  int       v8i;
typedef __attribute__((ext_vector_type(4)))  int       v4i;

union FragAB { v16bf v; uint32_t u[8]; };

// float -> bf16 round-to-nearest-even
__device__ __forceinline__ uint16_t f2bf(float x) {
  union { float f; uint32_t u; } c; c.f = x;
  uint32_t u = c.u;
  u += 0x7FFFu + ((u >> 16) & 1u);
  return (uint16_t)(u >> 16);
}

#if HAVE_TDM
// Issue a TDM load of a 64x64 bf16 tile (row length 64 elems = 32 dwords)
// into LDS at lds_off, inserting 3 pad dwords after every 32 data dwords so
// the LDS row stride is KP=35 dwords (bank-conflict-free fragment reads).
//   stride0_elems: global row stride in bf16 elements (64 for K, S_ for V^T)
__device__ __forceinline__ void tdm_load_tile(uint32_t lds_off, const void* gptr,
                                              uint32_t stride0_elems) {
  const uint64_t ga = (uint64_t)(uintptr_t)gptr;
  v4u g0;
  g0[0] = 1u;                                   // count=1, user descriptor
  g0[1] = lds_off;                              // lds_addr (bytes)
  g0[2] = (uint32_t)ga;                         // global_addr[31:0]
  g0[3] = (uint32_t)(ga >> 32) | (2u << 30);    // global_addr[56:32] | type=2
  v8i g1;
  //          pad_amount=3(code2)  pad_interval=32dw(code4)  pad_en  data_size=2B
  g1[0] = (int)((2u << 25)       | (4u << 22)             | (1u << 20) | (1u << 16));
  g1[1] = (int)(64u << 16);                     // tensor_dim0 = 64 (lo16 in [31:16])
  g1[2] = (int)(64u << 16);                     // tensor_dim1 = 64
  g1[3] = (int)(64u << 16);                     // tile_dim0 = 64
  g1[4] = 64;                                   // tile_dim1 = 64, tile_dim2 = 0
  g1[5] = (int)stride0_elems;                   // tensor_dim0_stride[31:0]
  g1[6] = 0;                                    // stride0[47:32]=0, stride1 unused
  g1[7] = 0;
  const v4i z4 = {0, 0, 0, 0};                  // 2-D tensor: groups 2/3 zero
  const v8i z8 = {0, 0, 0, 0, 0, 0, 0, 0};      // unused trailing group
  __builtin_amdgcn_tensor_load_to_lds(g0, g1, z4, z4, z8, 0);
}
#endif

// ---------------------------------------------------------------------------
// Phase 1: q = x@Wq+bq, k = x@Wk+bk, v = x@Wv+bv  (fp32 math, bf16 out)
// q,k stored row-major [B*S, H]; v stored TRANSPOSED [B, O, S] so the
// attention kernel reads Vt B-fragments as contiguous bf16 pairs along keys.
// ---------------------------------------------------------------------------
__global__ void __launch_bounds__(192) qkv_proj_kernel(
    const float* __restrict__ x,
    const float* __restrict__ Wq, const float* __restrict__ bq,
    const float* __restrict__ Wk, const float* __restrict__ bk,
    const float* __restrict__ Wv, const float* __restrict__ bv,
    uint16_t* __restrict__ qb, uint16_t* __restrict__ kb,
    uint16_t* __restrict__ vbT) {
  __shared__ float xs[E_];
  const int row = blockIdx.x;          // 0 .. B*S-1
  const int t   = threadIdx.x;         // 0 .. 191
  if (t < E_) xs[t] = x[(size_t)row * E_ + t];
  __syncthreads();

  const int grp = t >> 6;              // 0=q, 1=k, 2=v
  const int col = t & 63;
  const float* W    = (grp == 0) ? Wq : (grp == 1) ? Wk : Wv;
  const float* bias = (grp == 0) ? bq : (grp == 1) ? bk : bv;

  float acc = bias[col];
#pragma unroll 8
  for (int e = 0; e < E_; ++e) acc += xs[e] * W[e * 64 + col];

  const uint16_t h = f2bf(acc);
  if (grp == 0) {
    qb[(size_t)row * H_ + col] = h;
  } else if (grp == 1) {
    kb[(size_t)row * H_ + col] = h;
  } else {
    const int b = row >> 12, s = row & (S_ - 1);
    vbT[((size_t)b * O_ + col) * S_ + s] = h;
  }
}

// ---------------------------------------------------------------------------
// Phase 2: flash attention. One wave = 16 query rows, full O=64.
// Double-buffered K/V tiles; TDM (tensor_load_to_lds) prefetches tile i+1
// while all 8 waves compute on tile i.
// ---------------------------------------------------------------------------
__global__ void __launch_bounds__(256) flash_attn_kernel(
    const uint16_t* __restrict__ qb, const uint16_t* __restrict__ kb,
    const uint16_t* __restrict__ vbT, float* __restrict__ out) {
  // per buffer: K tile (64*KP dwords) then V^T tile (64*KP dwords)
  __shared__ uint32_t KVt[2][KTILE * KP + O_ * KP];
  __shared__ uint16_t Pw[8][16 * PSTR];    // per-wave P scratch (C->A relayout)

  const int tid  = threadIdx.x;
  const int wave = tid >> 5;
  const int lane = tid & 31;
  const int n16  = lane & 15;
  const int hi   = lane >> 4;
  const int b    = blockIdx.y;
  const int q0   = blockIdx.x * QTILE + wave * 16;

  const uint32_t* qbp = (const uint32_t*)qb;

  // Q A-fragments: 2 chunks of K=32 covering H=64. A layout (16-bit):
  // lane row = lane&15; VGPR j holds K pair ((j>>2)<<4) + hi*8 + (j&3)*2.
  FragAB aq[2];
  {
    const size_t rowd = ((size_t)b * S_ + q0 + n16) * (H_ / 2);
#pragma unroll
    for (int c = 0; c < 2; ++c)
#pragma unroll
      for (int j = 0; j < 8; ++j) {
        const int kA = ((j >> 2) << 4) + (hi << 3) + ((j & 3) << 1) + 32 * c;
        aq[c].u[j] = qbp[rowd + (kA >> 1)];
      }
  }

  v8f acc[4];
#pragma unroll
  for (int ot = 0; ot < 4; ++ot) acc[ot] = (v8f){};
  float mrow[8], lrow[8];
#pragma unroll
  for (int r = 0; r < 8; ++r) { mrow[r] = -__builtin_inff(); lrow[r] = 0.0f; }

#if HAVE_TDM
  // Prologue: wave 0 issues the DMA for tile 0 into buffer 0.
  if (wave == 0) {
    tdm_load_tile((uint32_t)(uintptr_t)&KVt[0][0],
                  kb + ((size_t)b * S_) * H_, H_);
    tdm_load_tile((uint32_t)(uintptr_t)&KVt[0][KTILE * KP],
                  vbT + (size_t)b * O_ * S_, S_);
  }
#endif

  for (int it = 0; it < NIT; ++it) {
    const int cur = it & 1;
    const int key0 = it * KTILE;
    const uint32_t* Kt = &KVt[cur][0];
    const uint32_t* Vt = &KVt[cur][KTILE * KP];

#if HAVE_TDM
    if (wave == 0) __builtin_amdgcn_s_wait_tensorcnt(0);   // tile it landed
    __syncthreads();   // release all waves; prev compute on buf[cur^1] done
    if (wave == 0 && it + 1 < NIT) {
      const int knext = key0 + KTILE;
      tdm_load_tile((uint32_t)(uintptr_t)&KVt[cur ^ 1][0],
                    kb + ((size_t)b * S_ + knext) * H_, H_);
      tdm_load_tile((uint32_t)(uintptr_t)&KVt[cur ^ 1][KTILE * KP],
                    vbT + (size_t)b * O_ * S_ + knext, S_);
    }
#else
    // Fallback: cooperative load (256 threads, 8 dwords per tile each).
    {
      const uint32_t* kbp = (const uint32_t*)kb;
      const uint32_t* vbp = (const uint32_t*)vbT;
      const int r = tid >> 2, seg = tid & 3;
      const size_t kg = ((size_t)b * S_ + key0 + r) * (H_ / 2) + seg * 8;
#pragma unroll
      for (int j = 0; j < 8; ++j) KVt[cur][r * KP + seg * 8 + j] = kbp[kg + j];
      const size_t vg = (((size_t)b * O_ + r) * S_ + key0) / 2 + seg * 8;
#pragma unroll
      for (int j = 0; j < 8; ++j)
        KVt[cur][KTILE * KP + r * KP + seg * 8 + j] = vbp[vg + j];
    }
    __syncthreads();
#endif

    // ---- scores S = Q K^T : 16 x 64, fp32 ----
    v8f sc[4];
#pragma unroll
    for (int t = 0; t < 4; ++t) {
      v8f s = (v8f){};
#pragma unroll
      for (int c = 0; c < 2; ++c) {
        FragAB bk;
#pragma unroll
        for (int j = 0; j < 8; ++j) {
          const int kB = ((j >> 2) << 4) + (hi << 3) + ((j & 3) << 1) + 32 * c;   // B layout K pair
          bk.u[j] = Kt[(t * 16 + n16) * KP + (kB >> 1)];
        }
        s = __builtin_amdgcn_wmma_f32_16x16x32_bf16(
                false, aq[c].v, false, bk.v, (short)0, s, false, false);
      }
      sc[t] = s;
    }

    // ---- online softmax (row stats per C-layout row r, halves independent) --
    float scale[8];
#pragma unroll
    for (int r = 0; r < 8; ++r) {
      float mx = fmaxf(fmaxf(sc[0][r], sc[1][r]), fmaxf(sc[2][r], sc[3][r]));
#pragma unroll
      for (int d = 1; d < 16; d <<= 1) mx = fmaxf(mx, __shfl_xor(mx, d, 32));
      const float mnew = fmaxf(mrow[r], mx);
      scale[r] = __expf(mrow[r] - mnew);
      float rs = 0.0f;
#pragma unroll
      for (int t = 0; t < 4; ++t) {
        const float p = __expf(sc[t][r] - mnew);
        sc[t][r] = p;
        rs += p;
      }
#pragma unroll
      for (int d = 1; d < 16; d <<= 1) rs += __shfl_xor(rs, d, 32);
      lrow[r] = lrow[r] * scale[r] + rs;
      mrow[r] = mnew;
#pragma unroll
      for (int ot = 0; ot < 4; ++ot) acc[ot][r] *= scale[r];
    }

    // ---- P: C-layout f32 -> bf16 row-major LDS (wave-private, no barrier) ---
#pragma unroll
    for (int t = 0; t < 4; ++t)
#pragma unroll
      for (int r = 0; r < 8; ++r) {
        const int m = r + 8 * hi;
        const int n = t * 16 + n16;
        Pw[wave][m * PSTR + n] = f2bf(sc[t][r]);
      }

    // ---- O += P @ V ----
#pragma unroll
    for (int c = 0; c < 2; ++c) {
      FragAB pa;
#pragma unroll
      for (int j = 0; j < 8; ++j) {
        const int kA = ((j >> 2) << 4) + (hi << 3) + ((j & 3) << 1) + 32 * c;
        pa.u[j] = *(const uint32_t*)&Pw[wave][n16 * PSTR + kA];
      }
#pragma unroll
      for (int ot = 0; ot < 4; ++ot) {
        FragAB bv;
#pragma unroll
        for (int j = 0; j < 8; ++j) {
          const int kB = ((j >> 2) << 4) + (hi << 3) + ((j & 3) << 1) + 32 * c;
          bv.u[j] = Vt[(ot * 16 + n16) * KP + (kB >> 1)];
        }
        acc[ot] = __builtin_amdgcn_wmma_f32_16x16x32_bf16(
                      false, pa.v, false, bv.v, (short)0, acc[ot], false, false);
      }
    }
#if !HAVE_TDM
    __syncthreads();   // before next tile overwrites the buffer
#endif
  }

  // ---- epilogue: divide by softmax denominator, store fp32 ----
#pragma unroll
  for (int ot = 0; ot < 4; ++ot)
#pragma unroll
    for (int r = 0; r < 8; ++r) {
      const int m = r + 8 * hi;
      const int o = ot * 16 + n16;
      out[((size_t)b * S_ + q0 + m) * O_ + o] = acc[ot][r] / lrow[r];
    }
}

// ---------------------------------------------------------------------------
extern "C" void kernel_launch(void* const* d_in, const int* in_sizes, int n_in,
                              void* d_out, int out_size, void* d_ws, size_t ws_size,
                              hipStream_t stream) {
  (void)in_sizes; (void)n_in; (void)out_size; (void)ws_size;
  const float* x  = (const float*)d_in[0];
  const float* Wq = (const float*)d_in[1];
  const float* bq = (const float*)d_in[2];
  const float* Wk = (const float*)d_in[3];
  const float* bk = (const float*)d_in[4];
  const float* Wv = (const float*)d_in[5];
  const float* bv = (const float*)d_in[6];

  uint16_t* qb  = (uint16_t*)d_ws;                       // [B*S, H] bf16
  uint16_t* kb  = qb + (size_t)B_ * S_ * H_;             // [B*S, H] bf16
  uint16_t* vbT = kb + (size_t)B_ * S_ * H_;             // [B, O, S] bf16

  qkv_proj_kernel<<<B_ * S_, 192, 0, stream>>>(x, Wq, bq, Wk, bk, Wv, bv,
                                               qb, kb, vbT);
  flash_attn_kernel<<<dim3(S_ / QTILE, B_), 256, 0, stream>>>(
      qb, kb, vbT, (float*)d_out);
}
